// DIDAModuleD4_19542101196954
// MI455X (gfx1250) — matrix operand order repaired and checked by reading, not compiled, not run on hardware
//
#include <hip/hip_runtime.h>
#include <hip/hip_bf16.h>

// ---------------------------------------------------------------------------
// MI455X (gfx1250) implementation.
//
// Pipeline (all on `stream`, serialized):
//   1. mean_kernel      : mean_x[32,256] = mean over HW of x
//   2. prep_kernel      : g = relu(conv_w @ mean_x + b); dynamic taps kers[2048][43]
//   3. cvt_convw_kernel : conv_w -> bf16
//   4. wcat_kernel      : Wcat[384,192] = fc_w_blk @ fuse_w (bf16), b' = fc_b + fc_w@fuse_b~
//   5. f_gemm_kernel    : f = relu(conv_w @ x)   (WMMA bf16, f32 accum)
//   6. depthwise_kernel : o1/o2/o3 (5x5 d1, 3x3 d2, 3x3 d4) -> ocat bf16 [32][192][4096]
//                         (f plane staged to LDS via GLOBAL_LOAD_ASYNC_TO_LDS_B128)
//   7. out_gemm_kernel  : out = Wcat @ ocat + b' (WMMA bf16, f32 accum)
// ---------------------------------------------------------------------------

typedef __attribute__((ext_vector_type(8)))  float          v8f;
typedef __attribute__((ext_vector_type(16))) __bf16         v16bf;
typedef __attribute__((ext_vector_type(16))) unsigned short v16u;

union BF16Frag {
    v16u  u;
    v16bf b;
    uint4 q[2];
};

__device__ __forceinline__ unsigned short f2bf(float f) {
    union { float f; unsigned u; } v;
    v.f = f;
    unsigned u = v.u;
    u += 0x7FFFu + ((u >> 16) & 1u);   // round-to-nearest-even
    return (unsigned short)(u >> 16);
}

// ---------------------------------------------------------------------------
// 1. Per-(n,c) spatial mean of x:  mean_x[n*256+c]
// ---------------------------------------------------------------------------
__global__ void mean_kernel(const float* __restrict__ x, float* __restrict__ mean_x) {
    __shared__ float red[256];
    const int nc = blockIdx.x;                       // 0..8191
    const float* p = x + (size_t)nc * 4096;
    float s = 0.0f;
    for (int i = threadIdx.x; i < 4096; i += 256) s += p[i];
    red[threadIdx.x] = s;
    __syncthreads();
    for (int off = 128; off > 0; off >>= 1) {
        if (threadIdx.x < off) red[threadIdx.x] += red[threadIdx.x + off];
        __syncthreads();
    }
    if (threadIdx.x == 0) mean_x[nc] = red[0] * (1.0f / 4096.0f);
}

// ---------------------------------------------------------------------------
// 2. g[n,cm] = relu(conv_w @ mean_x + conv_b); dynamic depthwise taps (43/chan)
//    taps layout: [0..24] = 5x5 d1, [25..33] = 3x3 d2, [34..42] = 3x3 d4
// ---------------------------------------------------------------------------
__global__ void prep_kernel(const float* __restrict__ mean_x,
                            const float* __restrict__ conv_w,
                            const float* __restrict__ conv_b,
                            const float* __restrict__ ck_w,  const float* __restrict__ ck_b,
                            const float* __restrict__ ck2_w, const float* __restrict__ ck2_b,
                            const float* __restrict__ ckd4_w,const float* __restrict__ ckd4_b,
                            const float* __restrict__ kern_w, const float* __restrict__ kern_b,
                            const float* __restrict__ kern2_w,const float* __restrict__ kern2_b,
                            const float* __restrict__ kernd4_w,const float* __restrict__ kernd4_b,
                            float* __restrict__ kers /* [32*64][43] */) {
    const int t = blockIdx.x * blockDim.x + threadIdx.x;
    if (t >= 32 * 64) return;
    const int n = t >> 6, cm = t & 63;
    float s = conv_b[cm];
    const float* w  = conv_w + (size_t)cm * 256;
    const float* mx = mean_x + (size_t)n * 256;
    for (int c = 0; c < 256; ++c) s += w[c] * mx[c];
    const float g = fmaxf(s, 0.0f);
    float* kp = kers + (size_t)t * 43;
    float sw = ck_w[0], sb = ck_b[0];
    for (int i = 0; i < 25; ++i) kp[i]      = sw * (g * kern_w[i]   + kern_b[i])   + sb;
    sw = ck2_w[0];  sb = ck2_b[0];
    for (int i = 0; i < 9;  ++i) kp[25 + i] = sw * (g * kern2_w[i]  + kern2_b[i])  + sb;
    sw = ckd4_w[0]; sb = ckd4_b[0];
    for (int i = 0; i < 9;  ++i) kp[34 + i] = sw * (g * kernd4_w[i] + kernd4_b[i]) + sb;
}

// ---------------------------------------------------------------------------
// 3. conv_w (64x256) -> bf16
// ---------------------------------------------------------------------------
__global__ void cvt_convw_kernel(const float* __restrict__ conv_w,
                                 unsigned short* __restrict__ out) {
    const int i = blockIdx.x * 256 + threadIdx.x;
    if (i < 64 * 256) out[i] = f2bf(conv_w[i]);
}

// ---------------------------------------------------------------------------
// 4. Wcat[o, i*64+c] = sum_p fc_w[o, i*128+p] * fuse_w[p, c]   (bf16 out)
//    bprime[o] = fc_b[o] + sum_{q<384} fc_w[o,q] * fuse_b[q % 128]
// ---------------------------------------------------------------------------
__global__ void wcat_kernel(const float* __restrict__ fc_w,   // [384][384]
                            const float* __restrict__ fuse_w, // [128][64]
                            const float* __restrict__ fc_b,   // [384]
                            const float* __restrict__ fuse_b, // [128]
                            unsigned short* __restrict__ wbf, // [384][192]
                            float* __restrict__ bprime) {     // [384]
    const int t = blockIdx.x * blockDim.x + threadIdx.x;
    if (t >= 384 * 192) return;
    const int o = t / 192, j = t % 192;
    const int i = j >> 6, c = j & 63;
    const float* fw = fc_w + (size_t)o * 384 + i * 128;
    float s = 0.0f;
    for (int p = 0; p < 128; ++p) s += fw[p] * fuse_w[p * 64 + c];
    wbf[t] = f2bf(s);
    if (j == 0) {
        float b = fc_b[o];
        const float* fo = fc_w + (size_t)o * 384;
        for (int p = 0; p < 384; ++p) b += fo[p] * fuse_b[p & 127];
        bprime[o] = b;
    }
}

// ---------------------------------------------------------------------------
// 5. f-GEMM: f[n][m][col] = relu( sum_k conv_w[m][k] * x[n][k][col] + b[m] )
//    M=64, K=256, cols=4096/n-plane.  Block = 4 waves, tile = 64 rows x 64 cols.
//    B staged in LDS transposed to [col][k] so each lane's WMMA B fragment is
//    16 contiguous bf16 (two uint4 LDS reads), in the contiguous k order this code uses for both A and B:
//      B frag lane l: n = l%16, k = (l/16)*16 + e  (e = 0..15)
//      A frag lane l: m = l%16, k = half*16 + {0..15}
// ---------------------------------------------------------------------------
#define FG_LDB 264   // K=256 + 8 pad (ushorts per staged column)

__global__ void f_gemm_kernel(const float* __restrict__ x,            // [32][256][4096]
                              const unsigned short* __restrict__ Abf, // [64][256] bf16
                              const float* __restrict__ bias,         // [64]
                              float* __restrict__ f) {                // [32][64][4096]
    __shared__ __align__(16) unsigned short Bs[64 * FG_LDB];
    const int n    = blockIdx.y;
    const int col0 = blockIdx.x * 64;
    const float* xb = x + (size_t)n * 256 * 4096;
    const int tid = threadIdx.x;

    // Stage B tile: coalesced float4 reads along cols, bf16 scatter into [col][k].
    {
        const int cg = tid & 15;          // 4-col group
        const int kr = tid >> 4;          // 8 k-rows per pass
        for (int k = kr; k < 256; k += 8) {
            if (k + 64 < 256)             // pull later K-rows toward GL2 early
                __builtin_prefetch(xb + (size_t)(k + 64) * 4096 + col0 + cg * 4, 0, 0);
            const float4 v = *(const float4*)(xb + (size_t)k * 4096 + col0 + cg * 4);
            const int cb = cg * 4;
            Bs[(cb + 0) * FG_LDB + k] = f2bf(v.x);
            Bs[(cb + 1) * FG_LDB + k] = f2bf(v.y);
            Bs[(cb + 2) * FG_LDB + k] = f2bf(v.z);
            Bs[(cb + 3) * FG_LDB + k] = f2bf(v.w);
        }
    }
    __syncthreads();

    const int wave = tid >> 5, lane = tid & 31;
    const int half = lane >> 4, lm = lane & 15;
    const int wcol = wave * 16;           // this wave's 16 columns in the tile

    v8f acc[4] = {};
    for (int kb = 0; kb < 256; kb += 32) {
        BF16Frag bf;
        const unsigned short* bp = &Bs[(wcol + lm) * FG_LDB + kb + half * 16];
        bf.q[0] = *(const uint4*)(bp);
        bf.q[1] = *(const uint4*)(bp + 8);
#pragma unroll
        for (int mt = 0; mt < 4; ++mt) {
            BF16Frag af;
            const unsigned short* ap = Abf + (size_t)(mt * 16 + lm) * 256 + kb + half * 16;
            af.q[0] = *(const uint4*)(ap);
            af.q[1] = *(const uint4*)(ap + 8);
            acc[mt] = __builtin_amdgcn_wmma_f32_16x16x32_bf16(
                false, af.b, false, bf.b, (short)0, acc[mt], false, false);
        }
    }

    const int col = col0 + wcol + lm;
#pragma unroll
    for (int mt = 0; mt < 4; ++mt) {
#pragma unroll
        for (int r = 0; r < 8; ++r) {
            const int m = mt * 16 + r + half * 8;   // C/D layout: VGPR r -> M = r + 8*half
            f[((size_t)n * 64 + m) * 4096 + col] = fmaxf(acc[mt][r] + bias[m], 0.0f);
        }
    }
}

// ---------------------------------------------------------------------------
// 6. Dynamic depthwise convs -> ocat bf16 [32][192][4096]
//    One block per (n, cm) plane; LDS-tiled 72x72 plane with halo 4 (zero pad).
//    The 64x64 fp32 interior is staged with GLOBAL_LOAD_ASYNC_TO_LDS_B128:
//    per-lane 16B global chunk -> 16B LDS chunk, no VGPR round trip (ASYNCcnt).
//    Tap offsets (cross-correlation like lax.conv): 5x5 d1 -> +-2; 3x3 d2 pad2
//    -> {-2,0,2}; 3x3 d4 pad4 -> {-4,0,4}.
// ---------------------------------------------------------------------------
__global__ void depthwise_kernel(const float* __restrict__ f,
                                 const float* __restrict__ kers,
                                 unsigned short* __restrict__ ocat) {
    __shared__ __align__(16) float plane[72 * 72];
    __shared__ float taps[43];
    const int ncm = blockIdx.x;                 // 0..2047
    const int n = ncm >> 6, cm = ncm & 63;
    const float* fp = f + (size_t)ncm * 4096;
    const int tid = threadIdx.x;

    // Zero the whole plane (covers the halo), then barrier before async writes
    // land so the zero-stores cannot race the async interior fill.
    for (int i = tid; i < 72 * 72; i += 256) plane[i] = 0.0f;
    if (tid < 43) taps[tid] = kers[(size_t)ncm * 43 + tid];
    __syncthreads();

    // Async-copy the 64x64 interior: 64 rows x 256B = 1024 chunks of 16B.
    {
        const unsigned planeBase = (unsigned)(uintptr_t)(void*)plane;
#pragma unroll
        for (int i = 0; i < 4; ++i) {
            const int q = i * 256 + tid;          // 0..1023
            const int y = q >> 4;                 // source row 0..63
            const int xb = (q & 15) * 16;         // byte offset within row
            const unsigned lds_addr = planeBase + (unsigned)((y + 4) * 72 + 4) * 4u + (unsigned)xb;
            const char* gaddr = (const char*)fp + (size_t)y * 256 + xb;
            asm volatile("global_load_async_to_lds_b128 %0, %1, off"
                         :: "v"(lds_addr), "v"(gaddr)
                         : "memory");
        }
        asm volatile("s_wait_asynccnt 0x0" ::: "memory");
    }
    __syncthreads();

    unsigned short* o1 = ocat + (size_t)n * 192 * 4096 + (size_t)cm * 4096;
    unsigned short* o2 = o1 + (size_t)64 * 4096;
    unsigned short* o3 = o1 + (size_t)128 * 4096;
    for (int p = tid; p < 4096; p += 256) {
        const int h = p >> 6, w = p & 63;
        const float* c = &plane[(h + 4) * 72 + (w + 4)];
        float s1 = 0.0f, s2 = 0.0f, s3 = 0.0f;
#pragma unroll
        for (int dy = -2; dy <= 2; ++dy)
#pragma unroll
            for (int dx = -2; dx <= 2; ++dx)
                s1 += taps[(dy + 2) * 5 + (dx + 2)] * c[dy * 72 + dx];
#pragma unroll
        for (int dy = -1; dy <= 1; ++dy)
#pragma unroll
            for (int dx = -1; dx <= 1; ++dx) {
                const float t2 = taps[25 + (dy + 1) * 3 + (dx + 1)];
                const float t3 = taps[34 + (dy + 1) * 3 + (dx + 1)];
                s2 += t2 * c[(dy * 2) * 72 + dx * 2];
                s3 += t3 * c[(dy * 4) * 72 + dx * 4];
            }
        o1[p] = f2bf(s1);
        o2[p] = f2bf(s2);
        o3[p] = f2bf(s3);
    }
}

// ---------------------------------------------------------------------------
// 7. out-GEMM: out[n][m][col] = sum_k Wcat[m][k] * ocat[n][k][col] + b'[m]
//    M=384 (6 m-tiles per wave, 4 wave-column-groups), K=192, 64-col tile/block.
// ---------------------------------------------------------------------------
#define OG_LDB 200   // K=192 + 8 pad

__global__ void out_gemm_kernel(const unsigned short* __restrict__ ocat, // [32][192][4096]
                                const unsigned short* __restrict__ Wbf,  // [384][192]
                                const float* __restrict__ bprime,        // [384]
                                float* __restrict__ out) {               // [32][384][4096]
    __shared__ __align__(16) unsigned short Bs[64 * OG_LDB];
    const int n    = blockIdx.z;
    const int mg   = blockIdx.y;                 // m base = mg*96
    const int col0 = blockIdx.x * 64;
    const unsigned short* ob = ocat + (size_t)n * 192 * 4096;
    const int tid = threadIdx.x;

    // Stage B tile: uint4 = 8 bf16 along cols, scatter into [col][k].
    {
        const int cg = tid & 7;          // 8-col group
        const int kr = tid >> 3;         // 16 k-rows per pass
        for (int k = kr; k < 192; k += 16) {
            if (k + 64 < 192)
                __builtin_prefetch(ob + (size_t)(k + 64) * 4096 + col0 + cg * 8, 0, 0);
            const uint4 v = *(const uint4*)(ob + (size_t)k * 4096 + col0 + cg * 8);
            const unsigned short* s = (const unsigned short*)&v;
#pragma unroll
            for (int j = 0; j < 8; ++j) Bs[(cg * 8 + j) * OG_LDB + k] = s[j];
        }
    }
    __syncthreads();

    const int wave = tid >> 5, lane = tid & 31;
    const int half = lane >> 4, lm = lane & 15;
    const int wcol = wave * 16;

    v8f acc[6] = {};
    for (int kb = 0; kb < 192; kb += 32) {
        BF16Frag bf;
        const unsigned short* bp = &Bs[(wcol + lm) * OG_LDB + kb + half * 16];
        bf.q[0] = *(const uint4*)(bp);
        bf.q[1] = *(const uint4*)(bp + 8);
#pragma unroll
        for (int mt = 0; mt < 6; ++mt) {
            BF16Frag af;
            const int m = mg * 96 + mt * 16 + lm;
            const unsigned short* ap = Wbf + (size_t)m * 192 + kb + half * 16;
            af.q[0] = *(const uint4*)(ap);
            af.q[1] = *(const uint4*)(ap + 8);
            acc[mt] = __builtin_amdgcn_wmma_f32_16x16x32_bf16(
                false, af.b, false, bf.b, (short)0, acc[mt], false, false);
        }
    }

    const int col = col0 + wcol + lm;
#pragma unroll
    for (int mt = 0; mt < 6; ++mt) {
#pragma unroll
        for (int r = 0; r < 8; ++r) {
            const int m = mg * 96 + mt * 16 + r + half * 8;
            out[((size_t)n * 384 + m) * 4096 + col] = acc[mt][r] + bprime[m];
        }
    }
}

// ---------------------------------------------------------------------------
// Host launcher
// ---------------------------------------------------------------------------
extern "C" void kernel_launch(void* const* d_in, const int* in_sizes, int n_in,
                              void* d_out, int out_size, void* d_ws, size_t ws_size,
                              hipStream_t stream) {
    (void)in_sizes; (void)n_in; (void)out_size; (void)ws_size;

    const float* x        = (const float*)d_in[0];
    const float* conv_w   = (const float*)d_in[1];
    const float* conv_b   = (const float*)d_in[2];
    const float* ck_w     = (const float*)d_in[3];
    const float* ck_b     = (const float*)d_in[4];
    const float* ck2_w    = (const float*)d_in[5];
    const float* ck2_b    = (const float*)d_in[6];
    const float* ckd4_w   = (const float*)d_in[7];
    const float* ckd4_b   = (const float*)d_in[8];
    const float* kern_w   = (const float*)d_in[9];
    const float* kern_b   = (const float*)d_in[10];
    const float* kern2_w  = (const float*)d_in[11];
    const float* kern2_b  = (const float*)d_in[12];
    const float* kernd4_w = (const float*)d_in[13];
    const float* kernd4_b = (const float*)d_in[14];
    const float* fuse_w   = (const float*)d_in[15];
    const float* fuse_b   = (const float*)d_in[16];
    const float* fc_w     = (const float*)d_in[17];
    const float* fc_b     = (const float*)d_in[18];
    float* out = (float*)d_out;

    // Workspace layout (256B aligned slices)
    char* ws = (char*)d_ws;
    size_t off = 0;
    auto take = [&](size_t bytes) -> char* {
        char* p = ws + off;
        off = (off + bytes + 255) & ~(size_t)255;
        return p;
    };
    float*          mean_x  = (float*)take((size_t)32 * 256 * 4);
    float*          kers    = (float*)take((size_t)32 * 64 * 43 * 4);
    unsigned short* convwbf = (unsigned short*)take((size_t)64 * 256 * 2);
    unsigned short* wcatbf  = (unsigned short*)take((size_t)384 * 192 * 2);
    float*          bprime  = (float*)take((size_t)384 * 4);
    float*          fbuf    = (float*)take((size_t)32 * 64 * 4096 * 4);
    unsigned short* ocat    = (unsigned short*)take((size_t)32 * 192 * 4096 * 2);

    mean_kernel<<<32 * 256, 256, 0, stream>>>(x, mean_x);
    prep_kernel<<<8, 256, 0, stream>>>(mean_x, conv_w, conv_b,
                                       ck_w, ck_b, ck2_w, ck2_b, ckd4_w, ckd4_b,
                                       kern_w, kern_b, kern2_w, kern2_b,
                                       kernd4_w, kernd4_b, kers);
    cvt_convw_kernel<<<64, 256, 0, stream>>>(conv_w, convwbf);
    wcat_kernel<<<(384 * 192 + 255) / 256, 256, 0, stream>>>(fc_w, fuse_w, fc_b, fuse_b,
                                                             wcatbf, bprime);
    f_gemm_kernel<<<dim3(64, 32), 128, 0, stream>>>(x, convwbf, conv_b, fbuf);
    depthwise_kernel<<<32 * 64, 256, 0, stream>>>(fbuf, kers, ocat);
    out_gemm_kernel<<<dim3(64, 4, 32), 128, 0, stream>>>(ocat, wcatbf, bprime, out);
}
